// Temporal_Transformer_46273977647576
// MI455X (gfx1250) — matrix operand order repaired and checked by reading, not compiled, not run on hardware
//
#include <hip/hip_runtime.h>

#define B_ 16
#define C_ 64
#define T_ 2048
#define D_ 32
#define SBLK 128

typedef __attribute__((ext_vector_type(16))) __bf16 v16bf;
typedef __attribute__((ext_vector_type(8)))  __bf16 v8bf;
typedef __attribute__((ext_vector_type(8)))  float  v8f;

// ---- WMMA helper (CDNA5 wave32, 16x16x32 bf16 -> f32) -----------------------
__device__ __forceinline__ v8f wmma_bf16(v16bf a, v16bf b, v8f c) {
  return __builtin_amdgcn_wmma_f32_16x16x32_bf16(false, a, false, b, (short)0, c,
                                                 false, false);
}

// A-operand: row-major [M][ldk] bf16 (global or LDS).
// lanes 0-15: K in [kb,kb+8)∪[kb+16,kb+24); lanes 16-31: [kb+8,kb+16)∪[kb+24,kb+32).
__device__ __forceinline__ v16bf load_A_rm(const __bf16* base, int ldk, int kb, int lane) {
  const int hl = (lane >> 4) & 1, m = lane & 15;
  const __bf16* row = base + m * ldk;
  v8bf lo = *(const v8bf*)(row + kb + hl * 8);
  v8bf hi = *(const v8bf*)(row + kb + 16 + hl * 8);
  return __builtin_shufflevector(lo, hi, 0,1,2,3,4,5,6,7,8,9,10,11,12,13,14,15);
}

// B-operand: B is KxN staged column-major as row-major [N][ldk].
// lanes 0-15: N=lane, K in [kb,kb+8)∪[kb+16,kb+24); lanes 16-31: N=lane-16, K in [kb+8,kb+16)∪[kb+24,kb+32).
__device__ __forceinline__ v16bf load_B_rm(const __bf16* base, int ldk, int nbase,
                                           int kb, int lane) {
  const int hl = (lane >> 4) & 1, n = lane & 15;
  const __bf16* row = base + (nbase + n) * ldk + kb + hl * 8;
  v8bf lo = *(const v8bf*)(row);
  v8bf hi = *(const v8bf*)(row + 16);
  return __builtin_shufflevector(lo, hi, 0,1,2,3,4,5,6,7,8,9,10,11,12,13,14,15);
}

// B-operand straight from f32 weights in global (converted inline).
__device__ __forceinline__ v16bf load_B_f32(const float* base, int ldk, int nbase,
                                            int kb, int lane) {
  const int hl = (lane >> 4) & 1, n = lane & 15;
  const float* row = base + (nbase + n) * ldk + kb + hl * 8;
  v16bf out;
#pragma unroll
  for (int i = 0; i < 16; ++i) out[i] = (__bf16)row[i < 8 ? i : i + 8];
  return out;
}

// Online column max/sumexp; column s lives in lane pair (s, s+16) of C/D layout.
__device__ __forceinline__ void online_update(const v8f vals, float& m, float& l) {
  float mx = vals[0];
#pragma unroll
  for (int r = 1; r < 8; ++r) mx = fmaxf(mx, vals[r]);
  mx = fmaxf(mx, __shfl_xor(mx, 16, 32));
  const float mnew = fmaxf(m, mx);
  float s = 0.f;
#pragma unroll
  for (int r = 0; r < 8; ++r) s += __expf(vals[r] - mnew);
  s += __shfl_xor(s, 16, 32);
  l = l * __expf(m - mnew) + s;
  m = mnew;
}

// ---- prep 0: zero Y accumulator --------------------------------------------
__global__ void zero_kernel(float* __restrict__ p, int n) {
  int i = blockIdx.x * 256 + threadIdx.x;
  if (i < n) p[i] = 0.f;
}

// ---- prep 1: x [b][c][t] f32 -> xbfT [b][t][c] bf16 (LDS tile transpose) ----
__global__ __launch_bounds__(256)
void transpose_bf16_kernel(const float* __restrict__ x, __bf16* __restrict__ xbfT) {
  __shared__ __align__(16) __bf16 tile[64][72];
  const int b = blockIdx.x / (T_ / 64);
  const int t0 = (blockIdx.x % (T_ / 64)) * 64;
  const float* xb = x + (size_t)b * C_ * T_;
  for (int idx = threadIdx.x; idx < 64 * 64; idx += 256) {
    int tt = idx & 63, c = idx >> 6;           // reads coalesced along t
    tile[tt][c] = (__bf16)xb[c * T_ + t0 + tt];
  }
  __syncthreads();
  __bf16* ob = xbfT + ((size_t)b * T_ + t0) * C_;
  for (int idx = threadIdx.x; idx < 64 * 64; idx += 256) {
    int cc = idx & 63, tt = idx >> 6;          // writes coalesced along c
    ob[tt * C_ + cc] = tile[tt][cc];
  }
}

// ---- prep 2: q/k/v projections via WMMA; v stored pre-transposed [d][t] -----
__global__ __launch_bounds__(192)
void qkv_kernel(const __bf16* __restrict__ xbfT,
                const float* __restrict__ wq, const float* __restrict__ bq,
                const float* __restrict__ wk, const float* __restrict__ bk,
                const float* __restrict__ wv, const float* __restrict__ bv,
                __bf16* __restrict__ qT, __bf16* __restrict__ kT,
                __bf16* __restrict__ vTt) {
  const int b = blockIdx.x / (T_ / 16);
  const int t0 = (blockIdx.x % (T_ / 16)) * 16;
  const int wave = threadIdx.x >> 5, lane = threadIdx.x & 31;
  const int hl = lane >> 4, nn = lane & 15;
  const int mat = wave >> 1, dh = wave & 1;    // 6 waves: {q,k,v} x {d-half}
  const float* w    = (mat == 0) ? wq : (mat == 1) ? wk : wv;
  const float* bias = (mat == 0) ? bq : (mat == 1) ? bk : bv;
  const __bf16* At = xbfT + ((size_t)b * T_ + t0) * C_;
  v8f acc = {};
  acc = wmma_bf16(load_A_rm(At, C_, 0, lane),  load_B_f32(w, C_, dh * 16, 0, lane),  acc);
  acc = wmma_bf16(load_A_rm(At, C_, 32, lane), load_B_f32(w, C_, dh * 16, 32, lane), acc);
  const int d = dh * 16 + nn;
  const float bb = bias[d];
  if (mat == 2) {
    __bf16* vb = vTt + ((size_t)b * D_ + d) * T_;
#pragma unroll
    for (int r = 0; r < 8; ++r) vb[t0 + r + 8 * hl] = (__bf16)(acc[r] + bb);
  } else {
    __bf16* ob = ((mat == 0) ? qT : kT) + ((size_t)b * T_ + t0) * D_;
#pragma unroll
    for (int r = 0; r < 8; ++r) ob[(r + 8 * hl) * D_ + d] = (__bf16)(acc[r] + bb);
  }
}

// ---- main fused kernel: one WG per (batch, 128-column s-block) --------------
__global__ __launch_bounds__(256)
void moca_attn_kernel(const __bf16* __restrict__ xbfT, const __bf16* __restrict__ qT,
                      const __bf16* __restrict__ kT, const __bf16* __restrict__ vTt,
                      const float* __restrict__ w2d, const float* __restrict__ b2d,
                      float* __restrict__ Yws) {
  __shared__ __align__(16) __bf16 Vbuf[D_][136];   // v block [d][s]
  __shared__ __align__(16) __bf16 Pbuf[64][136];   // exp(u) chunk [t][s]

  const int b  = blockIdx.x / (T_ / SBLK);
  const int s0 = (blockIdx.x % (T_ / SBLK)) * SBLK;
  const int tid = threadIdx.x, wave = tid >> 5, lane = tid & 31;
  const int hl = lane >> 4, nn = lane & 15;
  const int sw = wave * 16;                        // this wave's s-stripe
  const float w0 = w2d[0], w1 = w2d[1], b0 = b2d[0];

  const __bf16* xB = xbfT + (size_t)b * T_ * C_;
  const __bf16* qB = qT + (size_t)b * T_ * D_;
  const __bf16* kB = kT + (size_t)b * T_ * D_;

  // stage V block (contiguous row copies thanks to [d][t] layout)
  {
    const __bf16* vB = vTt + (size_t)b * D_ * T_ + s0;
    for (int idx = tid; idx < D_ * SBLK; idx += 256) {
      int d = idx >> 7, j = idx & 127;
      Vbuf[d][j] = vB[d * T_ + j];
    }
  }
  // loop-invariant B-operands (held in registers for the whole kernel)
  const v16bf Bg0 = load_B_rm(xB, C_, s0 + sw, 0, lane);
  const v16bf Bg1 = load_B_rm(xB, C_, s0 + sw, 32, lane);
  const v16bf Ba  = load_B_rm(kB, D_, s0 + sw, 0, lane);
  __syncthreads();

  auto tile_ga = [&](int t0, v8f& gg, v8f& aa) {
    v8f g = {};
    g = wmma_bf16(load_A_rm(xB + (size_t)t0 * C_, C_, 0, lane),  Bg0, g);
    g = wmma_bf16(load_A_rm(xB + (size_t)t0 * C_, C_, 32, lane), Bg1, g);
    gg = g;
    v8f ac = {};
    ac = wmma_bf16(load_A_rm(qB + (size_t)t0 * D_, D_, 0, lane), Ba, ac);
    const float isd = 0.17677669529663687f;  // 1/sqrt(32)
#pragma unroll
    for (int r = 0; r < 8; ++r) aa[r] = ac[r] * isd;
  };
  auto prefetch_next = [&](int t0) {
    if (t0 + 16 < T_) {
      __builtin_prefetch((const char*)(xB + (size_t)(t0 + 16) * C_) + lane * 64, 0, 3);
      __builtin_prefetch((const char*)(qB + (size_t)(t0 + 16) * D_) + lane * 32, 0, 3);
    }
  };

  // ---- pass 1: online column max/sumexp for G and A (no barriers) -----------
  float m_g = -3e38f, l_g = 0.f, m_a = -3e38f, l_a = 0.f;
  for (int it = 0; it < T_ / 16; ++it) {
    const int t0 = it * 16;
    prefetch_next(t0);
    v8f gg, aa;
    tile_ga(t0, gg, aa);
    online_update(gg, m_g, l_g);
    online_update(aa, m_a, l_a);
  }
  const float rl_g = 1.f / l_g, rl_a = 1.f / l_a;

  // ---- pass 2: column sum of exp(u) (no barriers) ---------------------------
  float lu_part = 0.f;
  for (int it = 0; it < T_ / 16; ++it) {
    const int t0 = it * 16;
    prefetch_next(t0);
    v8f gg, aa;
    tile_ga(t0, gg, aa);
#pragma unroll
    for (int r = 0; r < 8; ++r) {
      float u = w0 * (__expf(gg[r] - m_g) * rl_g) +
                w1 * (__expf(aa[r] - m_a) * rl_a) + b0;
      lu_part += __expf(u);
    }
  }
  const float l_u = lu_part + __shfl_xor(lu_part, 16, 32);
  const float rlu = 1.f / l_u;

  // fold 1/l_u into this wave's 16 V columns
  __syncthreads();
#pragma unroll
  for (int d = 0; d < 16; ++d) {
    int dd = hl * 16 + d;
    Vbuf[dd][sw + nn] = (__bf16)((float)Vbuf[dd][sw + nn] * rlu);
  }
  __syncthreads();

  // hoist this wave's Y B-operands (fixed d-half, 4 K-chunks)
  v16bf Bv[4];
#pragma unroll
  for (int kc = 0; kc < 4; ++kc)
    Bv[kc] = load_B_rm(&Vbuf[0][0], 136, (wave & 1) * 16, kc * 32, lane);

  // ---- pass 3: P = exp(u) in 64-row chunks, Y += P·V^T (full-K per wave) ----
  for (int ch = 0; ch < T_ / 64; ++ch) {
#pragma unroll
    for (int sub = 0; sub < 4; ++sub) {
      const int t0 = ch * 64 + sub * 16;
      prefetch_next(t0);
      v8f gg, aa;
      tile_ga(t0, gg, aa);
#pragma unroll
      for (int r = 0; r < 8; ++r) {
        float u = w0 * (__expf(gg[r] - m_g) * rl_g) +
                  w1 * (__expf(aa[r] - m_a) * rl_a) + b0;
        Pbuf[sub * 16 + r + 8 * hl][sw + nn] = (__bf16)__expf(u);
      }
    }
    __syncthreads();
    {
      const int tsub = wave >> 1, dh = wave & 1;  // 8 waves = 4 t-subtiles x 2 d-halves
      v8f y = {};
#pragma unroll
      for (int kc = 0; kc < 4; ++kc)
        y = wmma_bf16(load_A_rm(&Pbuf[tsub * 16][0], 136, kc * 32, lane), Bv[kc], y);
      float* Yb = Yws + ((size_t)b * T_ + ch * 64 + tsub * 16) * D_;
#pragma unroll
      for (int r = 0; r < 8; ++r)
        atomicAdd(&Yb[(r + 8 * hl) * D_ + dh * 16 + nn], y[r]);
      (void)dh;
    }
    __syncthreads();
  }
}

// ---- epilogue: out[b,c,t] = x + wz(64x32) @ Y[b,t,:] + bz -------------------
__global__ __launch_bounds__(128)
void proj_kernel(const float* __restrict__ Yws, const float* __restrict__ wz,
                 const float* __restrict__ bz, const float* __restrict__ x,
                 float* __restrict__ out) {
  const int bidx = blockIdx.x;
  const int b = bidx / (T_ / 16);
  const int t0 = (bidx % (T_ / 16)) * 16;
  const int wave = threadIdx.x >> 5, lane = threadIdx.x & 31;
  const int hl = lane >> 4, nn = lane & 15;

  const float* Yt = Yws + ((size_t)b * T_ + t0) * D_;
  v16bf A;
  {
    const int m = lane & 15;
    const int k0 = hl * 8, k1 = 16 + hl * 8;
#pragma unroll
    for (int i = 0; i < 8; ++i) A[i] = (__bf16)Yt[m * D_ + k0 + i];
#pragma unroll
    for (int i = 0; i < 8; ++i) A[8 + i] = (__bf16)Yt[m * D_ + k1 + i];
  }
  const int ct = wave;  // 4 waves = 4 c-tiles
  v16bf Bv = load_B_f32(wz, D_, ct * 16, 0, lane);
  v8f acc = {};
  acc = wmma_bf16(A, Bv, acc);

  const float* xb = x + (size_t)b * C_ * T_;
  float* ob = out + (size_t)b * C_ * T_;
  const int c = ct * 16 + nn;
  const float bias = bz[c];
#pragma unroll
  for (int r = 0; r < 8; ++r) {
    const int t = t0 + r + 8 * hl;
    ob[c * T_ + t] = xb[c * T_ + t] + acc[r] + bias;
  }
}

extern "C" void kernel_launch(void* const* d_in, const int* in_sizes, int n_in,
                              void* d_out, int out_size, void* d_ws, size_t ws_size,
                              hipStream_t stream) {
  (void)in_sizes; (void)n_in; (void)out_size; (void)ws_size;
  const float* x   = (const float*)d_in[0];
  const float* wq  = (const float*)d_in[1];
  const float* bq  = (const float*)d_in[2];
  const float* wk  = (const float*)d_in[3];
  const float* bk  = (const float*)d_in[4];
  const float* wv  = (const float*)d_in[5];
  const float* bv  = (const float*)d_in[6];
  const float* w2d = (const float*)d_in[7];
  const float* b2d = (const float*)d_in[8];
  const float* wz  = (const float*)d_in[9];
  const float* bz  = (const float*)d_in[10];
  float* out = (float*)d_out;

  // workspace layout (14 MB total):
  //   [0,4M)   Yws   f32 [B][T][D]
  //   [4M,8M)  xbfT  bf16 [B][T][C]
  //   [8M,10M) qT    bf16 [B][T][D]
  //   [10M,12M)kT    bf16 [B][T][D]
  //   [12M,14M)vTt   bf16 [B][D][T]
  char* ws = (char*)d_ws;
  float*  Yws  = (float*)ws;
  __bf16* xbfT = (__bf16*)(ws + (size_t)4 * 1024 * 1024);
  __bf16* qTp  = (__bf16*)(ws + (size_t)8 * 1024 * 1024);
  __bf16* kTp  = (__bf16*)(ws + (size_t)10 * 1024 * 1024);
  __bf16* vTp  = (__bf16*)(ws + (size_t)12 * 1024 * 1024);

  const int nY = B_ * T_ * D_;
  zero_kernel<<<(nY + 255) / 256, 256, 0, stream>>>(Yws, nY);
  transpose_bf16_kernel<<<B_ * (T_ / 64), 256, 0, stream>>>(x, xbfT);
  qkv_kernel<<<B_ * (T_ / 16), 192, 0, stream>>>(xbfT, wq, bq, wk, bk, wv, bv,
                                                 qTp, kTp, vTp);
  moca_attn_kernel<<<B_ * (T_ / SBLK), 256, 0, stream>>>(xbfT, qTp, kTp, vTp,
                                                         w2d, b2d, Yws);
  proj_kernel<<<B_ * (T_ / 16), 128, 0, stream>>>(Yws, wz, bz, x, out);
}
